// CoordAttention_89653147337396
// MI455X (gfx1250) — matrix operand order repaired and checked by reading, not compiled, not run on hardware
//
#include <hip/hip_runtime.h>
#include <hip/hip_bf16.h>
#include <math.h>

typedef __attribute__((ext_vector_type(16))) _Float16 v16h;
typedef __attribute__((ext_vector_type(8)))  float    v8f;

#define BB 2
#define SS 32
#define NN 512
#define PP 32
#define EPSF 1e-6f

// ---------------------------------------------------------------------------
// Kernel 0: per-(b,s) CoordNorm statistics: COM (atom_only_mask) and
// factor = set_weights[s] / (mean_length + eps).
// grid = B*S blocks, 256 threads.
// ---------------------------------------------------------------------------
__global__ void coordnorm_stats_kernel(const float* __restrict__ coord,
                                       const float* __restrict__ node_mask,
                                       const float* __restrict__ atom_mask,
                                       const float* __restrict__ set_w,
                                       float* __restrict__ stats) {
    __shared__ float red[256];
    const int bs = blockIdx.x;          // b*S + s
    const int s  = bs & (SS - 1);
    const int t  = threadIdx.x;
    const float* cbase = coord + (size_t)bs * NN * 3;
    const float* nmrow = node_mask + (size_t)bs * NN;
    const float* amrow = atom_mask + (size_t)bs * NN;

    auto block_reduce = [&](float v) -> float {
        red[t] = v; __syncthreads();
        for (int o = 128; o > 0; o >>= 1) {
            if (t < o) red[t] += red[t + o];
            __syncthreads();
        }
        float r = red[0]; __syncthreads();
        return r;
    };

    float sx = 0.f, sy = 0.f, sz = 0.f, sm = 0.f;
    for (int n = t; n < NN; n += 256) {
        float a = amrow[n];
        sx += cbase[n * 3 + 0] * a;
        sy += cbase[n * 3 + 1] * a;
        sz += cbase[n * 3 + 2] * a;
        sm += a;
    }
    float tx = block_reduce(sx);
    float ty = block_reduce(sy);
    float tz = block_reduce(sz);
    float tm = block_reduce(sm);
    float denom = fmaxf(tm, 1.0f);
    float comx = tx / denom, comy = ty / denom, comz = tz / denom;

    float sl = 0.f, sn = 0.f;
    for (int n = t; n < NN; n += 256) {
        float nmv = nmrow[n];
        float cx = (cbase[n * 3 + 0] - comx) * nmv;
        float cy = (cbase[n * 3 + 1] - comy) * nmv;
        float cz = (cbase[n * 3 + 2] - comz) * nmv;
        float sq = cx * cx + cy * cy + cz * cz;
        sl += (sq > 0.f) ? sqrtf(sq) : 0.f;
        sn += nmv;
    }
    float tl = block_reduce(sl);
    float tn = block_reduce(sn);
    if (t == 0) {
        float scaled = tl / tn;                      // mean length per node
        float factor = set_w[s] / (scaled + EPSF);
        float* o = stats + bs * 4;
        o[0] = comx; o[1] = comy; o[2] = comz; o[3] = factor;
    }
}

// ---------------------------------------------------------------------------
// Kernel 1: proj[b,c,n,p] = sum_s cnorm[b,s,n,c] * W_proj[p,s]
// One wave (32 lanes) per (b, c, 16-row M tile); 2 WMMAs (N tiles) per wave.
// A (16x32 f16): lane = half*16 + m, elem e -> K = e + 8*half + (e>=8 ? 8 : 0)
// B (32x16 f16): lane = half*16 + col, elem e -> K = e + 8*half + (e>=8 ? 8 : 0)
// D (16x16 f32): lane = half*16 + col, vgpr r -> M = r + 8*half
// grid = B*3*32 blocks of 32 threads.
// ---------------------------------------------------------------------------
__global__ void proj_wmma_kernel(const float* __restrict__ coord,
                                 const float* __restrict__ node_mask,
                                 const float* __restrict__ stats,
                                 const float* __restrict__ W_proj,
                                 float* __restrict__ proj) {
    const int bid   = blockIdx.x;
    const int b     = bid / 96;
    const int rem   = bid % 96;
    const int cdim  = rem / 32;
    const int mtile = rem % 32;
    const int lane  = threadIdx.x;
    const int mrow  = lane & 15;
    const int half  = lane >> 4;
    const int n     = mtile * 16 + mrow;

    v16h a;
#pragma unroll
    for (int e = 0; e < 16; ++e) {
        int k = e + 8 * half + ((e >= 8) ? 8 : 0);     // s index
        const float* st = stats + (b * SS + k) * 4;
        float com    = st[cdim];
        float factor = st[3];
        float nmv    = node_mask[(b * SS + k) * NN + n];
        float val = (coord[((size_t)(b * SS + k) * NN + n) * 3 + cdim] - com)
                    * nmv * nmv * factor;
        a[e] = (_Float16)val;
    }

#pragma unroll
    for (int ntile = 0; ntile < 2; ++ntile) {
        v16h bf;
#pragma unroll
        for (int e = 0; e < 16; ++e) {
            int k   = e + 8 * half + ((e >= 8) ? 8 : 0); // s index
            int col = mrow + ntile * 16;               // p index
            bf[e] = (_Float16)W_proj[col * SS + k];    // W_proj[p,s]
        }
        v8f c = {};
        v8f d = __builtin_amdgcn_wmma_f32_16x16x32_f16(
            false, a, false, bf, (short)0, c, false, false);
#pragma unroll
        for (int r = 0; r < 8; ++r) {
            int mm = r + 8 * half;
            int p  = mrow + ntile * 16;
            proj[((size_t)(b * 3 + cdim) * NN + (mtile * 16 + mm)) * PP + p] = d[r];
        }
    }
}

// ---------------------------------------------------------------------------
// Kernel 2: adjacency-masked online softmax + weighted direction accumulation.
// One wave per (b, i); lane == p. Adjacency row staged in LDS; messages
// streamed once with non-temporal loads; masked j skipped entirely.
// grid = B*N/8 blocks of 256 threads (8 waves).
// ---------------------------------------------------------------------------
__global__ void attn_kernel(const float* __restrict__ messages,
                            const int*   __restrict__ adj,
                            const float* __restrict__ proj,
                            float* __restrict__ upd) {
    __shared__ int s_adj[8][NN];        // 16 KB
    const int wid  = threadIdx.x >> 5;
    const int lane = threadIdx.x & 31;  // p
    const int gw   = blockIdx.x * 8 + wid;
    const int b    = gw >> 9;
    const int i    = gw & (NN - 1);

    // stage adjacency row + degree
    const int* adjrow = adj + (size_t)(b * NN + i) * NN;
    int degp = 0;
    for (int t = lane; t < NN; t += 32) {
        int a = adjrow[t];
        s_adj[wid][t] = a;
        degp += a;
    }
#pragma unroll
    for (int o = 16; o > 0; o >>= 1) degp += __shfl_xor(degp, o, 32);
    const bool rowzero = (degp == 0);   // row with no neighbours -> mask = 0
    __syncthreads();

    const float* pbx = proj + (size_t)(b * 3 + 0) * NN * PP;
    const float* pby = proj + (size_t)(b * 3 + 1) * NN * PP;
    const float* pbz = proj + (size_t)(b * 3 + 2) * NN * PP;
    const float pix = pbx[i * PP + lane];
    const float piy = pby[i * PP + lane];
    const float piz = pbz[i * PP + lane];
    const float* msgrow = messages + (size_t)(b * NN + i) * NN * PP;

    float m = -INFINITY, l = 0.f;
    float ax = 0.f, ay = 0.f, az = 0.f, asq = 0.f;

    for (int j = 0; j < NN; ++j) {
        if (!rowzero && (s_adj[wid][j] == 0)) continue;   // masked: attn == 0
        float sv = __builtin_nontemporal_load(&msgrow[j * PP + lane]);
        float vx = pix - pbx[j * PP + lane];
        float vy = piy - pby[j * PP + lane];
        float vz = piz - pbz[j * PP + lane];
        float sq = vx * vx + vy * vy + vz * vz;
        float dist = (sq > 0.f) ? sqrtf(sq) : 0.f;
        float rinv = 1.f / (dist + EPSF);
        // online softmax update
        float nm    = fmaxf(m, sv);
        float scale = __expf(m - nm);     // exp(-inf) = 0 on first hit
        float e     = __expf(sv - nm);
        l   = l * scale + e;
        float er = e * rinv;
        ax  = ax * scale + er * vx;
        ay  = ay * scale + er * vy;
        az  = az * scale + er * vz;
        asq = asq * scale * scale + e * e;
        m   = nm;
    }

    float invl = 1.f / l;
    float w    = sqrtf(asq) * invl;       // ||attn||_2 over j
    float f    = invl * w;
    upd[((size_t)(b * 3 + 0) * NN + i) * PP + lane] = ax * f;
    upd[((size_t)(b * 3 + 1) * NN + i) * PP + lane] = ay * f;
    upd[((size_t)(b * 3 + 2) * NN + i) * PP + lane] = az * f;
}

// ---------------------------------------------------------------------------
// Kernel 3: out[b,s,n,c] = sum_p upd[b,c,n,p] * W_attn[s,p]   (WMMA)
// Same tiling as kernel 1: A = upd[b,c] (512x32), B[k=p][col=s] = W_attn[s,p].
// ---------------------------------------------------------------------------
__global__ void out_wmma_kernel(const float* __restrict__ upd,
                                const float* __restrict__ W_attn,
                                float* __restrict__ out) {
    const int bid   = blockIdx.x;
    const int b     = bid / 96;
    const int rem   = bid % 96;
    const int cdim  = rem / 32;
    const int mtile = rem % 32;
    const int lane  = threadIdx.x;
    const int mrow  = lane & 15;
    const int half  = lane >> 4;
    const int n     = mtile * 16 + mrow;

    v16h a;
#pragma unroll
    for (int e = 0; e < 16; ++e) {
        int k = e + 8 * half + ((e >= 8) ? 8 : 0);     // p index
        a[e] = (_Float16)upd[((size_t)(b * 3 + cdim) * NN + n) * PP + k];
    }

#pragma unroll
    for (int ntile = 0; ntile < 2; ++ntile) {
        v16h bf;
#pragma unroll
        for (int e = 0; e < 16; ++e) {
            int k   = e + 8 * half + ((e >= 8) ? 8 : 0); // p index
            int col = mrow + ntile * 16;               // s index
            bf[e] = (_Float16)W_attn[col * PP + k];    // W_attn[s,p]
        }
        v8f c = {};
        v8f d = __builtin_amdgcn_wmma_f32_16x16x32_f16(
            false, a, false, bf, (short)0, c, false, false);
#pragma unroll
        for (int r = 0; r < 8; ++r) {
            int mm = mtile * 16 + r + 8 * half;        // n
            int sc = mrow + ntile * 16;                // s
            out[((size_t)(b * SS + sc) * NN + mm) * 3 + cdim] = d[r];
        }
    }
}

// ---------------------------------------------------------------------------
extern "C" void kernel_launch(void* const* d_in, const int* in_sizes, int n_in,
                              void* d_out, int out_size, void* d_ws, size_t ws_size,
                              hipStream_t stream) {
    const float* coord     = (const float*)d_in[0];   // [B,S,N,3]
    const float* messages  = (const float*)d_in[1];   // [B,N,N,P]
    const int*   adj       = (const int*)  d_in[2];   // [B,N,N]
    const float* node_mask = (const float*)d_in[3];   // [B,S,N]
    const float* atom_mask = (const float*)d_in[4];   // [B,S,N]
    const float* set_w     = (const float*)d_in[5];   // [1,S,1,1]
    const float* W_proj    = (const float*)d_in[6];   // [P,S]
    const float* W_attn    = (const float*)d_in[7];   // [S,P]
    float* out = (float*)d_out;                        // [B,S,N,3]

    char* ws = (char*)d_ws;
    float* stats = (float*)ws;                                  // B*S*4 floats
    float* proj  = (float*)(ws + 1024);                         // B*3*N*P floats
    float* upd   = (float*)(ws + 1024 + (size_t)BB * 3 * NN * PP * 4);

    coordnorm_stats_kernel<<<BB * SS, 256, 0, stream>>>(coord, node_mask,
                                                        atom_mask, set_w, stats);
    proj_wmma_kernel<<<BB * 3 * (NN / 16), 32, 0, stream>>>(coord, node_mask,
                                                            stats, W_proj, proj);
    attn_kernel<<<(BB * NN) / 8, 256, 0, stream>>>(messages, adj, proj, upd);
    out_wmma_kernel<<<BB * 3 * (NN / 16), 32, 0, stream>>>(upd, W_attn, out);
}
